// DifferentiableLattice_9887014715483
// MI455X (gfx1250) — matrix operand order repaired and checked by reading, not compiled, not run on hardware
//
#include <hip/hip_runtime.h>
#include <hip/hip_bf16.h>
#include <math.h>

// ---------------------------------------------------------------------------
// DifferentiableLattice on MI455X (gfx1250, wave32, WMMA).
// 9 serialized GEMM steps (16384x512 @ 512x512) in bf16 WMMA with f32 accum.
// State ping-pongs in workspace as bf16; output accumulated in f32.
// ---------------------------------------------------------------------------

typedef __attribute__((ext_vector_type(16))) __bf16 v16bf;
typedef __attribute__((ext_vector_type(8)))  __bf16 v8bf;
typedef __attribute__((ext_vector_type(8)))  float  v8f;

#define N_CELLS   512
#define BATCH     16384
#define MAX_STEPS 10

// workspace byte offsets (state buffers ~33.5MB total; L2 is 192MB so the
// working set is fully L2-resident)
#define W_OFF     0u                                   // 16 f32 softmax weights
#define DECAY_OFF 64u                                  // 1 f32
#define MAX_OFF   128u                                 // 16 f32 per-step global max
#define AF_OFF    256u                                 // 512 f32 angle factors
#define PROPT_OFF 4096u                                // 512*512 bf16, propT[n][k]=prop[n][k]
#define ST0_OFF   (1u << 20)                           // 16384*512 bf16 (16MB)
#define ST1_OFF   ((1u << 20) + (unsigned)BATCH * N_CELLS * 2u)

// ---------------------------------------------------------------------------
// Kernel 1: scalars — softmax(step_weights[:len]), decay clip, angle factors,
// zero max slots. One block of 512 threads.
// ---------------------------------------------------------------------------
__global__ void lattice_prep_scalars(const float* __restrict__ bounce,
                                     const float* __restrict__ sw,
                                     const float* __restrict__ decay_rate,
                                     const int*   __restrict__ n_steps_p,
                                     char* __restrict__ ws) {
  float* w   = (float*)(ws + W_OFF);
  float* dec = (float*)(ws + DECAY_OFF);
  float* mx  = (float*)(ws + MAX_OFF);
  float* af  = (float*)(ws + AF_OFF);
  const int t = threadIdx.x;

  if (t < N_CELLS) {
    float m = 0.f;
#pragma unroll
    for (int a = 0; a < 6; ++a) {
      float v = bounce[t * 6 + a];
      v = fminf(fmaxf(v, 0.f), 2.f);
      m += v;
    }
    m *= (1.0f / 6.0f);
    af[t] = 0.5f + 0.5f * cosf(m);
  }
  if (t < 16) mx[t] = 0.f;
  if (t == 0) {
    // clip(decay, min=2.0, max=0.99) saturates to 0.99
    *dec = fminf(fmaxf(decay_rate[0], 2.0f), 0.99f);
    int steps = *n_steps_p;
    if (steps > MAX_STEPS) steps = MAX_STEPS;
    const int len = steps + 1;           // len(history)
    float mm = -1e30f;
    for (int i = 0; i < len; ++i) mm = fmaxf(mm, sw[i]);
    float e[MAX_STEPS + 1];
    float s = 0.f;
    for (int i = 0; i < len; ++i) { e[i] = expf(sw[i] - mm); s += e[i]; }
    const float inv = 1.f / s;
    for (int i = 0; i < 16; ++i) w[i] = (i < len) ? e[i] * inv : 0.f;
  }
}

// ---------------------------------------------------------------------------
// Kernel 2: build prop row n, normalize, store bf16 row n untransposed, propT[n][k].
// Grid: 512 blocks x 256 threads (2 k's per thread).
// ---------------------------------------------------------------------------
__global__ void lattice_prep_prop(const float* __restrict__ adj,
                                  const float* __restrict__ stdv,
                                  const float* __restrict__ spv,
                                  const float* __restrict__ jpv,
                                  char* __restrict__ ws) {
  __shared__ float red[256];
  __shared__ float rowsum;
  const int n = blockIdx.x;
  const int t = threadIdx.x;

  const float s_dev    = fmaxf(fabsf(stdv[n]), 2.0f);
  const float straight = erff(0.5f / (s_dev * 1.41421356237f));
  const float sp       = fminf(fmaxf(spv[n], 0.f), 1.f);
  const float self_ret = straight * 0.3f * (1.f - sp * 0.5f);
  const float spread   = 1.f - straight + sp * 0.3f;

  float v0, v1;
  {
    int k = t;
    float jp = fminf(fmaxf(jpv[k], 0.f), 1.f);
    v0 = ((k == n) ? self_ret : 0.f) +
         adj[n * N_CELLS + k] * spread * (1.f + jp * 0.5f) * 0.7f;
    k = t + 256;
    jp = fminf(fmaxf(jpv[k], 0.f), 1.f);
    v1 = ((k == n) ? self_ret : 0.f) +
         adj[n * N_CELLS + k] * spread * (1.f + jp * 0.5f) * 0.7f;
  }
  red[t] = v0 + v1;
  __syncthreads();
  for (int off = 128; off > 0; off >>= 1) {
    if (t < off) red[t] += red[t + off];
    __syncthreads();
  }
  if (t == 0) rowsum = fmaxf(red[0], 1e-6f);
  __syncthreads();
  const float inv = 1.f / rowsum;

  __bf16* propT = (__bf16*)(ws + PROPT_OFF);
  propT[n * N_CELLS + t]         = (__bf16)(v0 * inv);
  propT[n * N_CELLS + t + 256]   = (__bf16)(v1 * inv);
}

// ---------------------------------------------------------------------------
// Kernel 3: out = w0*x, state0 = bf16(x), global max of step-0 state.
// ---------------------------------------------------------------------------
__global__ void lattice_init(const float* __restrict__ x0,
                             float* __restrict__ out,
                             char* __restrict__ ws) {
  __shared__ float red[256];
  const int i = blockIdx.x * 256 + threadIdx.x;
  const float* w  = (const float*)(ws + W_OFF);
  float* mx       = (float*)(ws + MAX_OFF);
  __bf16* st0     = (__bf16*)(ws + ST0_OFF);

  const float x = x0[i];
  out[i] = w[0] * x;
  st0[i] = (__bf16)x;

  red[threadIdx.x] = x;
  __syncthreads();
  for (int off = 128; off > 0; off >>= 1) {
    if (threadIdx.x < off)
      red[threadIdx.x] = fmaxf(red[threadIdx.x], red[threadIdx.x + off]);
    __syncthreads();
  }
  if (threadIdx.x == 0)
    atomicMax((int*)&mx[0], __float_as_int(red[0]));  // values are >= 0
}

// ---------------------------------------------------------------------------
// Kernel 4: one lattice step.
// Block = 256 threads = 8 waves. Block covers 16 batch rows x full N=512.
// Wave w computes a 16x64 output strip: 4 accumulators, K-loop of 16 x
// v_wmma_f32_16x16x32_bf16 per accumulator.
//
// A layout (16-bit A 16x32, per ISA): lane<16 -> M=lane, elems 0..7 = K kb+0..7,
// elems 8..15 = K kb+16..23; lane>=16 -> M=lane-16, K halves +8 / +24.
// B layout (dense 32x16): lane L holds column N=n0+L%16, elements = K in A's order, i.e.
// two 16B slices of propT row n: K kb+khalf+0..7 and +16..23.
// Epilogue fuses angle_factor*decay, sigmoid*max, bf16 state store,
// weighted accumulation into out, and the next-step global max.
// ---------------------------------------------------------------------------
__global__ void __launch_bounds__(256)
lattice_step(const int* __restrict__ n_steps_p, int s,
             float* __restrict__ out, char* __restrict__ ws) {
  int n_steps = *n_steps_p;
  if (n_steps > MAX_STEPS) n_steps = MAX_STEPS;
  if (s >= n_steps) return;   // uniform: EXEC stays all-ones for WMMA

  const float* w     = (const float*)(ws + W_OFF);
  const float  dec   = *(const float*)(ws + DECAY_OFF);
  float*       mx    = (float*)(ws + MAX_OFF);
  const float* af    = (const float*)(ws + AF_OFF);
  const __bf16* propT = (const __bf16*)(ws + PROPT_OFF);
  const __bf16* src   = (const __bf16*)(ws + ((s & 1) ? ST1_OFF : ST0_OFF));
  __bf16*       dst   = (__bf16*)(ws + ((s & 1) ? ST0_OFF : ST1_OFF));

  const int lane = threadIdx.x & 31;
  const int wave = threadIdx.x >> 5;
  const int m0   = blockIdx.x * 16;
  const int nb   = wave * 64;

  const float smax = fmaxf(mx[s], 0.1f);
  const float wgt  = w[s + 1];

  v8f acc0 = {}, acc1 = {}, acc2 = {}, acc3 = {};

  const int row   = m0 + (lane & 15);
  const int khalf = (lane >> 4) ? 8 : 0;

  for (int kb = 0; kb < N_CELLS; kb += 32) {
    // A tile: two 16B chunks per lane, documented 16-bit A VGPR layout
    v8bf alo = *(const v8bf*)(src + row * N_CELLS + kb + khalf);
    v8bf ahi = *(const v8bf*)(src + row * N_CELLS + kb + khalf + 16);
    v16bf a = __builtin_shufflevector(alo, ahi,
        0, 1, 2, 3, 4, 5, 6, 7, 8, 9, 10, 11, 12, 13, 14, 15);

    // B tiles: lane = N column, K in same order as A (buffer holds prop[n][k])
    const __bf16* brow = propT + (nb + (lane & 15)) * N_CELLS + kb + khalf;
    v16bf b0 = __builtin_shufflevector(*(const v8bf*)(brow), *(const v8bf*)(brow + 16),
        0, 1, 2, 3, 4, 5, 6, 7, 8, 9, 10, 11, 12, 13, 14, 15);
    v16bf b1 = __builtin_shufflevector(*(const v8bf*)(brow + 16 * N_CELLS), *(const v8bf*)(brow + 16 * N_CELLS + 16),
        0, 1, 2, 3, 4, 5, 6, 7, 8, 9, 10, 11, 12, 13, 14, 15);
    v16bf b2 = __builtin_shufflevector(*(const v8bf*)(brow + 32 * N_CELLS), *(const v8bf*)(brow + 32 * N_CELLS + 16),
        0, 1, 2, 3, 4, 5, 6, 7, 8, 9, 10, 11, 12, 13, 14, 15);
    v16bf b3 = __builtin_shufflevector(*(const v8bf*)(brow + 48 * N_CELLS), *(const v8bf*)(brow + 48 * N_CELLS + 16),
        0, 1, 2, 3, 4, 5, 6, 7, 8, 9, 10, 11, 12, 13, 14, 15);

    acc0 = __builtin_amdgcn_wmma_f32_16x16x32_bf16(false, a, false, b0, (short)0, acc0, false, false);
    acc1 = __builtin_amdgcn_wmma_f32_16x16x32_bf16(false, a, false, b1, (short)0, acc1, false, false);
    acc2 = __builtin_amdgcn_wmma_f32_16x16x32_bf16(false, a, false, b2, (short)0, acc2, false, false);
    acc3 = __builtin_amdgcn_wmma_f32_16x16x32_bf16(false, a, false, b3, (short)0, acc3, false, false);
  }

  // Epilogue. C/D layout: lane<16 -> (M=e, N=lane); lane>=16 -> (M=8+e, N=lane-16)
  const int nloc = lane & 15;
  const int mb   = (lane >> 4) ? 8 : 0;
  float vmax = 0.f;

  auto epilogue = [&](v8f acc, int nt) {
    const int  n   = nt + nloc;
    const float fa = af[n] * dec;
#pragma unroll
    for (int e = 0; e < 8; ++e) {
      const int m = m0 + mb + e;
      const float g = acc[e] * fa;
      const float v = (1.f / (1.f + expf(1.f - 2.f * g))) * smax; // sigmoid(2g-1)*smax
      dst[m * N_CELLS + n] = (__bf16)v;
      out[m * N_CELLS + n] += wgt * v;
      vmax = fmaxf(vmax, v);
    }
  };
  epilogue(acc0, nb);
  epilogue(acc1, nb + 16);
  epilogue(acc2, nb + 32);
  epilogue(acc3, nb + 48);

  __shared__ float red[256];
  red[threadIdx.x] = vmax;
  __syncthreads();
  for (int off = 128; off > 0; off >>= 1) {
    if (threadIdx.x < off)
      red[threadIdx.x] = fmaxf(red[threadIdx.x], red[threadIdx.x + off]);
    __syncthreads();
  }
  if (threadIdx.x == 0)
    atomicMax((int*)&mx[s + 1], __float_as_int(red[0])); // v > 0 always
}

// ---------------------------------------------------------------------------
extern "C" void kernel_launch(void* const* d_in, const int* in_sizes, int n_in,
                              void* d_out, int out_size, void* d_ws, size_t ws_size,
                              hipStream_t stream) {
  const float* x0     = (const float*)d_in[0];  // (16384, 512)
  const float* adj    = (const float*)d_in[1];  // (512, 512)
  const float* stdv   = (const float*)d_in[2];  // (512,)
  const float* spv    = (const float*)d_in[3];  // (512,)
  const float* jpv    = (const float*)d_in[4];  // (512,)
  const float* bounce = (const float*)d_in[5];  // (512, 6)
  const float* sw     = (const float*)d_in[6];  // (10,)
  const float* dr     = (const float*)d_in[7];  // (1,)
  const int*   nst    = (const int*)d_in[8];    // scalar int
  float* out = (float*)d_out;
  char*  ws  = (char*)d_ws;

  hipLaunchKernelGGL(lattice_prep_scalars, dim3(1), dim3(512), 0, stream,
                     bounce, sw, dr, nst, ws);
  hipLaunchKernelGGL(lattice_prep_prop, dim3(N_CELLS), dim3(256), 0, stream,
                     adj, stdv, spv, jpv, ws);
  hipLaunchKernelGGL(lattice_init, dim3((BATCH * N_CELLS) / 256), dim3(256), 0,
                     stream, x0, out, ws);
  for (int s = 0; s < MAX_STEPS; ++s) {
    hipLaunchKernelGGL(lattice_step, dim3(BATCH / 16), dim3(256), 0, stream,
                       nst, s, out, ws);
  }
}
